// Net_SDE_30872224924037
// MI455X (gfx1250) — compile-verified, matrix operand order checked by reading, not run on hardware
//
#include <hip/hip_runtime.h>

// ---------------------------------------------------------------------------
// Neural-SDE Monte Carlo for MI455X (gfx1250).
// Persistent cooperative kernel: 128 WGs x 256 threads (8 wave32s), 1 path /
// thread. Hidden 64x64 layers run as v_wmma_f32_16x16x32_bf16 with bf16
// weights/activations staged in LDS (padded stride 66 for bank-conflict-free
// access). One device-wide split barrier per step for batch mean/std.
// Layer loop is NOT unrolled and sched_barriers fence layer/cell boundaries
// to keep the live VGPR set under 256 (no s_set_vgpr_msb traffic).
// ---------------------------------------------------------------------------

#define N_PATHS   32768
#define N_STEPS   128
#define WID       64
#define TPB       256
#define NWG       (N_PATHS / TPB)     // 128 workgroups
#define H_STEP    (1.0f / 128.0f)
#define RATE_R    0.025f
#define ASTRIDE   66                  // activation row stride (ushorts), 33 dwords
#define WSTRIDE   66                  // weight row stride (ushorts)

typedef __attribute__((ext_vector_type(16))) __bf16        v16bf;
typedef __attribute__((ext_vector_type(8)))  float         v8f;
typedef __attribute__((ext_vector_type(8)))  unsigned int  v8u;

union ABu { v8u u; v16bf b; };

static __device__ inline unsigned short f2bf(float f) {
    return __builtin_bit_cast(unsigned short, (__bf16)f);   // native cvt, RNE
}
static __device__ inline float bf2f(unsigned short u) {
    return __uint_as_float(((unsigned int)u) << 16);
}
static __device__ inline float softplus_f(float x) {
    return fmaxf(x, 0.0f) + log1pf(__expf(-fabsf(x)));
}

struct Params {
    const float* dW; const float* dB;
    const float* Wh[9]; const float* bh[9];       // [cell*3 + layer]
    const float* Wi[3]; const float* bi[3]; const float* Wo[3];
    const float* rWi; const float* rWo; const float* rbi;
    float* stats;            // [N_STEPS*4]  sumS, sumS2, sumV, sumV2
    float* pacc;             // [8*19]       9 calls, 9 puts, sumV per maturity
    unsigned int* bar;       // [N_STEPS]    monotonic barrier counters
};

__global__ __launch_bounds__(TPB) void sde_sim(Params P) {
    __shared__ unsigned short sWh[9][WID * WSTRIDE];   // ~74 KB bf16 weights
    __shared__ unsigned short hbuf[2][TPB * ASTRIDE];  // ~66 KB activations
    __shared__ float sxn[TPB * 2];                     // xS, xV per path
    __shared__ float sWiF[3][WID * 3];
    __shared__ float sbhF[9][WID];
    __shared__ float sbiF[3][WID];
    __shared__ float sWoF[3][WID];
    __shared__ float sred[32];

    const int tid  = threadIdx.x;
    const int path = blockIdx.x * TPB + tid;
    const int lane = tid & 31;
    const int wave = tid >> 5;
    const int lhi  = lane >> 4;          // lanes 16..31: +8 on K, +8 on M
    const int ln   = lane & 15;

    // ---- stage weights in LDS (bf16, padded stride, for WMMA operands) ----
    {
        const int dims[3] = {3, 2, 2};
        for (int m = 0; m < 9; ++m)
            for (int i = tid; i < WID * WID; i += TPB)
                sWh[m][(i >> 6) * WSTRIDE + (i & 63)] = f2bf(P.Wh[m][i]);
        for (int m = 0; m < 9; ++m)
            if (tid < WID) sbhF[m][tid] = P.bh[m][tid];
        for (int c = 0; c < 3; ++c) {
            const int n = WID * dims[c];
            for (int i = tid; i < n; i += TPB) sWiF[c][i] = P.Wi[c][i];
            if (tid < WID) { sbiF[c][tid] = P.bi[c][tid]; sWoF[c][tid] = P.Wo[c][tid]; }
        }
    }
    const float rwi0 = P.rWi[0], rwi1 = P.rWi[1];
    const float rwo0 = P.rWo[0], rwo1 = P.rWo[1];
    const float rbi0 = P.rbi[0], rbi1 = P.rbi[1];
    __syncthreads();

    // 16-bit A-operand 16x32 layout (ISA 7.12.2): lane group selects +8 on K,
    // element j -> K = (j&7) + (j>=8 ? 16 : 0); pairs are dword-contiguous.
    auto loadA = [&](int buf, int rowbase, int kc) -> v16bf {
        const int off = (rowbase + ln) * ASTRIDE + kc * 32 + lhi * 8;
        ABu u;
#pragma unroll
        for (int p = 0; p < 4; ++p) {
            u.u[p]     = *(const unsigned int*)&hbuf[buf][off + 2 * p];
            u.u[4 + p] = *(const unsigned int*)&hbuf[buf][off + 16 + 2 * p];
        }
        return u.b;
    };
    auto loadB = [&](int m, int nbase, int kc) -> v16bf {
        const int off = (nbase + ln) * WSTRIDE + kc * 32 + lhi * 8;
        ABu u;
#pragma unroll
        for (int p = 0; p < 4; ++p) {
            u.u[p]     = *(const unsigned int*)&sWh[m][off + 2 * p];
            u.u[4 + p] = *(const unsigned int*)&sWh[m][off + 16 + 2 * p];
        }
        return u.b;
    };

    // hout = relu(hin @ W^T + b); each wave owns its 32 rows -> no WG barrier.
    // All 8 B fragments held in registers and reused across both M tiles.
    auto hiddenLayer = [&](int bin, int bout, int m) {
        __builtin_amdgcn_sched_barrier(0);
        v16bf Bf[4][2];
#pragma unroll
        for (int nt = 0; nt < 4; ++nt) {
            Bf[nt][0] = loadB(m, nt * 16, 0);
            Bf[nt][1] = loadB(m, nt * 16, 1);
        }
#pragma unroll
        for (int mt = 0; mt < 2; ++mt) {
            const int rowbase = wave * 32 + mt * 16;
            v16bf a0 = loadA(bin, rowbase, 0);
            v16bf a1 = loadA(bin, rowbase, 1);
#pragma unroll
            for (int nt = 0; nt < 4; ++nt) {
                const float b = sbhF[m][nt * 16 + ln];
                v8f c;
#pragma unroll
                for (int r = 0; r < 8; ++r) c[r] = b;
                c = __builtin_amdgcn_wmma_f32_16x16x32_bf16(
                        false, a0, false, Bf[nt][0], (short)0, c, false, false);
                c = __builtin_amdgcn_wmma_f32_16x16x32_bf16(
                        false, a1, false, Bf[nt][1], (short)0, c, false, false);
#pragma unroll
                for (int r = 0; r < 8; ++r)
                    hbuf[bout][(rowbase + r + 8 * lhi) * ASTRIDE + nt * 16 + ln] =
                        f2bf(fmaxf(c[r], 0.0f));
            }
        }
        __builtin_amdgcn_sched_barrier(0);
    };

    float S = 100.0f, Vv = 0.04f;

    for (int step = 0; step < N_STEPS; ++step) {
        const float t = (float)step * H_STEP;

        // ---- batch stats: WG-level LDS reduction, then one device atomic ----
        if (tid < 4) sred[tid] = 0.0f;
        __syncthreads();
        atomicAdd(&sred[0], S);      atomicAdd(&sred[1], S * S);
        atomicAdd(&sred[2], Vv);     atomicAdd(&sred[3], Vv * Vv);
        __syncthreads();
        if (tid == 0) {
            float* st = P.stats + step * 4;
#pragma unroll
            for (int i = 0; i < 4; ++i) atomicAdd(&st[i], sred[i]);
            __threadfence();
            unsigned int* c = P.bar + step;
            __hip_atomic_fetch_add(c, 1u, __ATOMIC_RELEASE, __HIP_MEMORY_SCOPE_AGENT);
            while (__hip_atomic_load(c, __ATOMIC_ACQUIRE, __HIP_MEMORY_SCOPE_AGENT)
                   < (unsigned int)gridDim.x)
                __builtin_amdgcn_s_sleep(1);
#pragma unroll
            for (int i = 0; i < 4; ++i)
                sred[4 + i] = __hip_atomic_load(&st[i], __ATOMIC_RELAXED,
                                                __HIP_MEMORY_SCOPE_AGENT);
        }
        __syncthreads();

        const float inv_n = 1.0f / (float)N_PATHS;
        const float mS = sred[4] * inv_n, mV = sred[6] * inv_n;
        float varS = (sred[5] - (float)N_PATHS * mS * mS) / (float)(N_PATHS - 1);
        float varV = (sred[7] - (float)N_PATHS * mV * mV) / (float)(N_PATHS - 1);
        const float sS = varS > 0.0f ? sqrtf(varS) : 1.0f;
        const float sV = varV > 0.0f ? sqrtf(varV) : 1.0f;
        const float xS = (S - mS) / sS, xV = (Vv - mV) / sV;

        sxn[tid * 2]     = xS;
        sxn[tid * 2 + 1] = xV;

        // ---- one MLP cell: per-column input layer, 3 WMMA layers, output ----
        auto runCell = [&](int c, int m0, int dim, float xsum) -> float {
            __syncthreads();                          // sxn + prior buf reads
            {   // input layer: thread owns column (weights in regs), sweeps rows
                const int col = tid & 63;
                const float w0 = sWiF[c][col * dim + 0];
                const float w1 = sWiF[c][col * dim + 1];
                const float w2 = (dim == 3) ? sWiF[c][col * dim + 2] : 0.0f;
                const float bb = sbiF[c][col];
#pragma unroll 8
                for (int i = 0; i < 64; ++i) {
                    const int row = i * 4 + (tid >> 6);
                    const float xb = (dim == 3) ? sxn[row * 2] : sxn[row * 2 + 1];
                    const float xc = (dim == 3) ? sxn[row * 2 + 1] : 0.0f;
                    const float a  = bb + w0 * t + w1 * xb + w2 * xc;
                    hbuf[0][row * ASTRIDE + col] = f2bf(fmaxf(a, 0.0f));
                }
            }
            __syncthreads();                          // cross-wave input rows
            // buffers: 0 -> 1 -> 0 -> 1 ; single code copy of the WMMA body
#pragma clang loop unroll(disable)
            for (int l = 0; l < 3; ++l)
                hiddenLayer(l & 1, (l & 1) ^ 1, m0 + l);
            // output layer: own row, dword loads (2 bf16 each)
            float a = xsum;
            const int rb = tid * ASTRIDE;
#pragma unroll 8
            for (int k = 0; k < 32; ++k) {
                const unsigned int d = *(const unsigned int*)&hbuf[1][rb + 2 * k];
                a += bf2f((unsigned short)d)         * sWoF[c][2 * k]
                   + bf2f((unsigned short)(d >> 16)) * sWoF[c][2 * k + 1];
            }
            __builtin_amdgcn_sched_barrier(0);
            return a;
        };

        const float diff_o = runCell(0, 0, 3, t + xS + xV);
        const float dfv_o  = runCell(1, 3, 2, t + xV);
        const float drv_o  = runCell(2, 6, 2, t + xV);

        // rho net: width-2 scalar MLP of t (same for every path)
        const float rh0 = fmaxf(rwi0 * t + rbi0, 0.0f);
        const float rh1 = fmaxf(rwi1 * t + rbi1, 0.0f);
        const float rho = tanhf(rwo0 * rh0 + rwo1 * rh1 + t);

        // ---- Euler step ----
        const float dw = P.dW[path * N_STEPS + step];
        const float db = P.dB[path * N_STEPS + step];
        if (step + 1 < N_STEPS) {
            __builtin_prefetch(&P.dW[path * N_STEPS + step + 1], 0, 1);
            __builtin_prefetch(&P.dB[path * N_STEPS + step + 1], 0, 1);
        }
        S  = fmaxf(S * (1.0f + RATE_R * H_STEP) + softplus_f(diff_o) * dw, 0.0f);
        Vv = fabsf(Vv + drv_o * H_STEP +
                   softplus_f(dfv_o) * (sqrtf(fmaxf(1.0f - rho * rho, 0.0f)) * db
                                        + rho * dw));

        // ---- maturity reductions (every 16th step) ----
        if ((step & 15) == 15) {
            const int mi = step >> 4;
            if (tid < 19) sred[8 + tid] = 0.0f;
            __syncthreads();
            const float KS[9] = {80.f, 85.f, 90.f, 95.f, 100.f, 105.f, 110.f, 115.f, 120.f};
#pragma unroll
            for (int k = 0; k < 9; ++k) {
                atomicAdd(&sred[8 + k],     fmaxf(S - KS[k], 0.0f));
                atomicAdd(&sred[8 + 9 + k], fmaxf(KS[k] - S, 0.0f));
            }
            atomicAdd(&sred[8 + 18], Vv);
            __syncthreads();
            if (tid < 19) atomicAdd(&P.pacc[mi * 19 + tid], sred[8 + tid]);
            __syncthreads();
        }
    }
}

__global__ void ws_init(float* w, int n) {
    for (int i = threadIdx.x; i < n; i += blockDim.x) w[i] = 0.0f;
}

__global__ void finalize(const float* pacc, float* out) {
    const int i = threadIdx.x;
    const float inv_n = 1.0f / (float)N_PATHS;
    if (i < 144) {                                 // 72 calls then 72 puts
        const int half = i / 72, j = i % 72, mi = j / 9, k = j % 9;
        const float t = (float)(16 * (mi + 1) - 1) * H_STEP;
        out[i] = pacc[mi * 19 + half * 9 + k] * inv_n * __expf(-RATE_R * t);
    } else if (i < 152) {                          // forward variances
        const int mi = i - 144;
        out[i] = pacc[mi * 19 + 18] * inv_n;
    }
}

extern "C" void kernel_launch(void* const* d_in, const int* in_sizes, int n_in,
                              void* d_out, int out_size, void* d_ws, size_t ws_size,
                              hipStream_t stream) {
    (void)in_sizes; (void)n_in; (void)out_size; (void)ws_size;
    // Leaf order: dW, dB, then per cell dict sorted keys {Wh0..2, Wi, Wo,
    // bh0..2, bi}; cells in setup order diff_p, diffV_p, driftV_p; then
    // rho_p {Wi, Wo, bi}.
    Params P;
    P.dW = (const float*)d_in[0];
    P.dB = (const float*)d_in[1];
    const int base[3] = {2, 11, 20};               // diff, diffV, driftV
    for (int c = 0; c < 3; ++c) {
        for (int l = 0; l < 3; ++l) {
            P.Wh[c * 3 + l] = (const float*)d_in[base[c] + l];
            P.bh[c * 3 + l] = (const float*)d_in[base[c] + 5 + l];
        }
        P.Wi[c] = (const float*)d_in[base[c] + 3];
        P.Wo[c] = (const float*)d_in[base[c] + 4];
        P.bi[c] = (const float*)d_in[base[c] + 8];
    }
    P.rWi = (const float*)d_in[29];
    P.rWo = (const float*)d_in[30];
    P.rbi = (const float*)d_in[31];

    float* ws = (float*)d_ws;
    P.stats = ws;                                  // 512 floats
    P.pacc  = ws + 512;                            // 152 floats (pad to 160)
    P.bar   = (unsigned int*)(ws + 512 + 160);     // 128 counters

    hipLaunchKernelGGL(ws_init, dim3(1), dim3(256), 0, stream, ws, 512 + 160 + 128);
    hipLaunchKernelGGL(sde_sim, dim3(NWG), dim3(TPB), 0, stream, P);
    hipLaunchKernelGGL(finalize, dim3(1), dim3(256), 0, stream, P.pacc, (float*)d_out);
}
